// SelfMultiheadAttention_87866440942296
// MI455X (gfx1250) — matrix operand order repaired and checked by reading, not compiled, not run on hardware
//
#include <hip/hip_runtime.h>

// ---------------------------------------------------------------------------
// Self multihead attention for MI455X (gfx1250, wave32, WMMA bf16 16x16x32).
//   T=1024, B=4, E=1024, H=16, D=64, total heads N = B*H = 64
// Pipeline:
//   K0 cvt      :  one-shot f32 -> bf16 packing of query, W_in, W_out
//   K1 qkv_proj :  block-cooperative bf16 GEMM (A tile staged in LDS via
//                  global_load_async_to_lds_b128) -> Q*scale [n][t][d],
//                  K [n][s][d], V^T [n][d][s]
//   K2 attn     :  register-resident flash attention per (head, 16 t-rows);
//                  S^T tiles via WMMA, online softmax, PV^T via WMMA
//   K3 out_proj :  same GEMM structure -> f32 output (t,b,e)
// ---------------------------------------------------------------------------

#define TGT    1024
#define BSZ    4
#define EMBED  1024
#define NHEAD  16
#define HDIM   64
#define SCAL   0.125f   // 64^-0.5

typedef __attribute__((ext_vector_type(16))) __bf16 v16bf;
typedef __attribute__((ext_vector_type(8)))  float  v8f;
typedef __attribute__((ext_vector_type(4)))  int    v4i;

#define AS1 __attribute__((address_space(1)))
#define AS3 __attribute__((address_space(3)))

#if __has_builtin(__builtin_amdgcn_global_load_async_to_lds_b128) && \
    __has_builtin(__builtin_amdgcn_s_wait_asynccnt)
#define HAVE_ASYNC_LDS 1
#else
#define HAVE_ASYNC_LDS 0
#endif

union ABu {
    v16bf v;
    uint4 q[2];
    unsigned u[8];
};

// round-to-nearest-even f32 -> bf16 (packed pair / scalar)
__device__ __forceinline__ unsigned pk2(float lo, float hi) {
    unsigned a = __float_as_uint(lo);
    unsigned b = __float_as_uint(hi);
    a = (a + 0x7FFFu + ((a >> 16) & 1u)) >> 16;
    b = (b + 0x7FFFu + ((b >> 16) & 1u)) >> 16;
    return a | (b << 16);
}
__device__ __forceinline__ unsigned short f2bf(float x) {
    unsigned a = __float_as_uint(x);
    return (unsigned short)((a + 0x7FFFu + ((a >> 16) & 1u)) >> 16);
}

__device__ __forceinline__ void ld8f(const float* __restrict__ p, float* d) {
    float4 a = *(const float4*)p;
    float4 b = *(const float4*)(p + 4);
    d[0] = a.x; d[1] = a.y; d[2] = a.z; d[3] = a.w;
    d[4] = b.x; d[5] = b.y; d[6] = b.z; d[7] = b.w;
}
__device__ __forceinline__ void ld8i(const int* __restrict__ p, int* d) {
    int4 a = *(const int4*)p;
    int4 b = *(const int4*)(p + 4);
    d[0] = a.x; d[1] = a.y; d[2] = a.z; d[3] = a.w;
    d[4] = b.x; d[5] = b.y; d[6] = b.z; d[7] = b.w;
}

#define WMMA_BF16(A, B, C) \
    __builtin_amdgcn_wmma_f32_16x16x32_bf16(false, (A), false, (B), (short)0, (C), false, false)

// A-operand (16x32): elems 0-7 at p[0..7], elems 8-15 at p[16..23] (bf16 elems)
__device__ __forceinline__ v16bf load_a_bf16(const unsigned short* p) {
    ABu a;
    a.q[0] = *(const uint4*)(p);
    a.q[1] = *(const uint4*)(p + 16);
    return a.v;
}
// B-operand (32x16): 16 contiguous bf16 elems at p[0..15]
__device__ __forceinline__ v16bf load_b_bf16(const unsigned short* __restrict__ p) {
    ABu a;
    a.q[0] = *(const uint4*)(p);
    a.q[1] = *(const uint4*)(p + 8);
    return a.v;
}

// cooperative copy of a 32 KB A tile (16 rows x 1024 bf16) into LDS
__device__ __forceinline__ void stage_a_tile(const unsigned short* src,
                                             unsigned short* lds, int tid) {
#if HAVE_ASYNC_LDS
    // addrspace-qualified bases (addrspacecast first, then arithmetic in-AS)
    AS1 char* gbase = (AS1 char*)src;
    AS3 char* lbase = (AS3 char*)lds;
#pragma unroll
    for (int i = 0; i < 8; ++i) {
        const int off = tid * 16 + i * 4096;   // bytes
        __builtin_amdgcn_global_load_async_to_lds_b128(
            (AS1 v4i*)(gbase + off), (AS3 v4i*)(lbase + off), 0, 0);
    }
    __builtin_amdgcn_s_wait_asynccnt(0);
    __syncthreads();
#else
#pragma unroll
    for (int i = 0; i < 8; ++i) {
        const int off = tid * 8 + i * 2048;    // bf16 elements
        *(uint4*)(lds + off) = *(const uint4*)(src + off);
    }
    __syncthreads();
#endif
}

// ---------------------------------------------------------------------------
// Kernel 0: bulk f32 -> packed bf16 (8 elems / thread, b128 in, b128 out)
// ---------------------------------------------------------------------------
__global__ __launch_bounds__(256) void cvt_bf16_kernel(
    const float* __restrict__ src, unsigned* __restrict__ dst, int n8)
{
    int idx = blockIdx.x * blockDim.x + threadIdx.x;
    if (idx >= n8) return;
    const float4* p = (const float4*)src + (size_t)idx * 2;
    float4 a = p[0];
    float4 b = p[1];
    uint4 o;
    o.x = pk2(a.x, a.y); o.y = pk2(a.z, a.w);
    o.z = pk2(b.x, b.y); o.w = pk2(b.z, b.w);
    ((uint4*)dst)[idx] = o;
}

// ---------------------------------------------------------------------------
// Kernel 1: qkv = query @ W_in^T + b_in ; scatter into Q/K/V^T bf16 layouts.
// Block = 8 waves sharing one 16-row A tile (LDS); each wave owns 128 cols.
// Grid: 256 row-tiles x 3 col-groups = 768 blocks.
// ---------------------------------------------------------------------------
__global__ __launch_bounds__(256) void qkv_proj_kernel(
    const unsigned short* __restrict__ Qb,   // query bf16, row-major 4096x1024
    const unsigned short* __restrict__ Wb,   // W_in  bf16, row-major 3072x1024
    const float* __restrict__ bin,
    unsigned short* __restrict__ qws, unsigned short* __restrict__ kws,
    unsigned short* __restrict__ vws)
{
    __shared__ unsigned short Atile[16 * EMBED];   // 32 KB

    const int tid    = threadIdx.x;
    const int lane   = tid & 31;
    const int lane16 = lane & 15;
    const int kg     = lane >> 4;
    const int wave   = tid >> 5;            // 0..7
    const int mt     = blockIdx.x / 3;      // 0..255
    const int ng     = blockIdx.x % 3;      // 0..2
    const int r0     = mt * 16;
    const int e0     = ng * 1024 + wave * 128;

    stage_a_tile(Qb + (size_t)r0 * EMBED, Atile, tid);

    v8f C[8] = {};
    const unsigned short* arow = Atile + lane16 * EMBED;
    for (int k = 0; k < EMBED; k += 32) {
        v16bf a = load_b_bf16(arow + k + kg * 16);    // ds_load_b128 x2
#pragma unroll
        for (int j = 0; j < 8; ++j) {
            const unsigned short* bp =
                Wb + (size_t)(e0 + j * 16 + lane16) * EMBED + k + kg * 16;
            C[j] = WMMA_BF16(a, load_b_bf16(bp), C[j]);
        }
    }

    const int seg = e0 >> 10;   // 0=q, 1=k, 2=v (128-col tile never crosses)
#pragma unroll
    for (int j = 0; j < 8; ++j) {
        const int e    = e0 + j * 16 + lane16;
        const float bb = bin[e];
        const int e2   = e & 1023;
        const int h    = e2 >> 6;
        const int d    = e2 & 63;
#pragma unroll
        for (int i = 0; i < 8; ++i) {
            const int r = r0 + kg * 8 + i;
            const int t = r >> 2, b = r & 3;
            const int n = b * NHEAD + h;
            const float val = C[j][i] + bb;
            if (seg == 0)
                qws[((size_t)n * TGT + t) * HDIM + d] = f2bf(val * SCAL);
            else if (seg == 1)
                kws[((size_t)n * TGT + t) * HDIM + d] = f2bf(val);
            else
                vws[((size_t)n * HDIM + d) * TGT + t] = f2bf(val);
        }
    }
}

// ---------------------------------------------------------------------------
// Kernel 2: flash attention, one wave per (head n, 16-row t tile). 4096 waves.
// Computes S^T (s x t) tiles so softmax stats are per-lane; converts P to the
// PV B-operand with 4 shfl_xor(16) packed-bf16 exchanges (no LDS round trip).
// ---------------------------------------------------------------------------
__global__ __launch_bounds__(256) void attn_kernel(
    const unsigned short* __restrict__ qws,
    const unsigned short* __restrict__ kws,
    const unsigned short* __restrict__ vws,
    const float* __restrict__ attn_mask,
    const float* __restrict__ attn_bias,
    const int* __restrict__ kpm,
    unsigned short* __restrict__ attnws)
{
    const int lane   = threadIdx.x & 31;
    const int lane16 = lane & 15;
    const int kg     = lane >> 4;
    const int wid    = (blockIdx.x * blockDim.x + threadIdx.x) >> 5;
    const int n  = wid >> 6;            // head index 0..63
    const int t0 = (wid & 63) * 16;
    const int b  = n >> 4, h = n & 15;
    const int t  = t0 + lane16;         // this lane's t column

    // Q as B-operand, 2 k-steps over d (col t = lane16, K = kg*16 + 0..15)
    ABu qb0, qb1;
    const unsigned short* qrow = qws + ((size_t)n * TGT + t) * HDIM;
    qb0.q[0] = *(const uint4*)(qrow + kg * 16);
    qb0.q[1] = *(const uint4*)(qrow + kg * 16 + 8);
    qb1.q[0] = *(const uint4*)(qrow + 32 + kg * 16);
    qb1.q[1] = *(const uint4*)(qrow + 32 + kg * 16 + 8);

    v8f O[4] = {};                       // O^T accum: d rows (4 tiles) x t cols
    const float NEG_INF = -__builtin_inff();
    float m = NEG_INF, l = 0.0f;

    const float* mrow = attn_mask + (size_t)t * TGT;
    const float* brow = attn_bias + ((size_t)n * TGT + t) * TGT;
    const int*   kp   = kpm + b * TGT;

    for (int sc = 0; sc < TGT; sc += 32) {
        // ---- scores S^T: two 16-row tiles (s = sc..+15, sc+16..+31) ----
        v8f S0 = {}, S1 = {};
#pragma unroll
        for (int ks = 0; ks < 2; ++ks) {
            ABu a;
            const unsigned short* kr =
                kws + ((size_t)n * TGT + sc + lane16) * HDIM + ks * 32 + kg * 16;
            a.q[0] = *(const uint4*)(kr);
            a.q[1] = *(const uint4*)(kr + 8);
            S0 = WMMA_BF16(a.v, ks ? qb1.v : qb0.v, S0);
            const unsigned short* kr1 = kr + 16 * HDIM;
            a.q[0] = *(const uint4*)(kr1);
            a.q[1] = *(const uint4*)(kr1 + 8);
            S1 = WMMA_BF16(a.v, ks ? qb1.v : qb0.v, S1);
        }

        if (sc + 32 < TGT)  // prefetch next bias chunk (dominant HBM stream)
            __builtin_prefetch(brow + sc + 32 + kg * 8, 0, 1);

        // ---- mask + bias + key padding; element s = sc(+16) + kg*8 + i ----
        const int sb = sc + kg * 8;
        float am0[8], am1[8], ab0[8], ab1[8];
        int   kp0[8], kp1[8];
        ld8f(mrow + sb, am0); ld8f(mrow + sb + 16, am1);
        ld8f(brow + sb, ab0); ld8f(brow + sb + 16, ab1);
        ld8i(kp   + sb, kp0); ld8i(kp   + sb + 16, kp1);

        float s0v[8], s1v[8];
        float cmax = NEG_INF;
#pragma unroll
        for (int i = 0; i < 8; ++i) {
            s0v[i] = kp0[i] ? NEG_INF : (S0[i] + am0[i] + ab0[i]);
            s1v[i] = kp1[i] ? NEG_INF : (S1[i] + am1[i] + ab1[i]);
            cmax = fmaxf(cmax, fmaxf(s0v[i], s1v[i]));
        }
        cmax = fmaxf(cmax, __shfl_xor(cmax, 16, 32));

        const float mn = fmaxf(m, cmax);
        const float alpha = (m == NEG_INF) ? 0.0f : __expf(m - mn);
        const bool finite = (mn != NEG_INF);

        float p0[8], p1[8], rs = 0.0f;
#pragma unroll
        for (int i = 0; i < 8; ++i) {
            p0[i] = finite ? __expf(s0v[i] - mn) : 0.0f;
            p1[i] = finite ? __expf(s1v[i] - mn) : 0.0f;
            rs += p0[i] + p1[i];
        }
        rs += __shfl_xor(rs, 16, 32);
        l = l * alpha + rs;
        m = mn;
#pragma unroll
        for (int dt = 0; dt < 4; ++dt)
#pragma unroll
            for (int i = 0; i < 8; ++i) O[dt][i] *= alpha;

        // ---- pack P to bf16 and exchange halves between lane pairs ----
        unsigned pka[4], pkb[4];
#pragma unroll
        for (int j = 0; j < 4; ++j) {
            pka[j] = pk2(p0[2 * j], p0[2 * j + 1]);
            pkb[j] = pk2(p1[2 * j], p1[2 * j + 1]);
        }
        ABu pb;
#pragma unroll
        for (int j = 0; j < 4; ++j) {
            unsigned send = (kg == 0) ? pkb[j] : pka[j];
            unsigned got  = __shfl_xor(send, 16, 32);
            pb.u[j]     = (kg == 0) ? pka[j] : got;
            pb.u[4 + j] = (kg == 0) ? got    : pkb[j];
        }

        // ---- O^T += V^T(d x s) * P^T(s x t) ----
#pragma unroll
        for (int dt = 0; dt < 4; ++dt) {
            ABu a;
            const unsigned short* vr =
                vws + ((size_t)n * HDIM + dt * 16 + lane16) * TGT + sc + kg * 16;
            a.q[0] = *(const uint4*)(vr);
            a.q[1] = *(const uint4*)(vr + 8);
            O[dt] = WMMA_BF16(a.v, pb.v, O[dt]);
        }
    }

    const float inv = 1.0f / l;
#pragma unroll
    for (int dt = 0; dt < 4; ++dt)
#pragma unroll
        for (int i = 0; i < 8; ++i) {
            const int d = dt * 16 + kg * 8 + i;
            attnws[((size_t)t * BSZ + b) * EMBED + h * HDIM + d] =
                f2bf(O[dt][i] * inv);
        }
}

// ---------------------------------------------------------------------------
// Kernel 3: out = attn(bf16) @ W_out^T(bf16) + b_out -> f32.
// Block = 8 waves sharing one 16-row A tile (LDS); grid = 256 blocks.
// ---------------------------------------------------------------------------
__global__ __launch_bounds__(256) void out_proj_kernel(
    const unsigned short* __restrict__ A,     // attn bf16 4096x1024
    const unsigned short* __restrict__ Wb,    // W_out bf16 1024x1024
    const float* __restrict__ bout,
    float* __restrict__ out)
{
    __shared__ unsigned short Atile[16 * EMBED];   // 32 KB

    const int tid    = threadIdx.x;
    const int lane   = tid & 31;
    const int lane16 = lane & 15;
    const int kg     = lane >> 4;
    const int wave   = tid >> 5;          // 0..7
    const int r0     = blockIdx.x * 16;
    const int e0     = wave * 128;

    stage_a_tile(A + (size_t)r0 * EMBED, Atile, tid);

    v8f C[8] = {};
    const unsigned short* arow = Atile + lane16 * EMBED;
    for (int k = 0; k < EMBED; k += 32) {
        v16bf a = load_b_bf16(arow + k + kg * 16);    // ds_load_b128 x2
#pragma unroll
        for (int j = 0; j < 8; ++j) {
            const unsigned short* bp =
                Wb + (size_t)(e0 + j * 16 + lane16) * EMBED + k + kg * 16;
            C[j] = WMMA_BF16(a, load_b_bf16(bp), C[j]);
        }
    }

#pragma unroll
    for (int j = 0; j < 8; ++j) {
        const int e = e0 + j * 16 + lane16;
        const float bb = bout[e];
#pragma unroll
        for (int i = 0; i < 8; ++i) {
            const int r = r0 + kg * 8 + i;
            out[(size_t)r * EMBED + e] = C[j][i] + bb;
        }
    }
}

// ---------------------------------------------------------------------------
extern "C" void kernel_launch(void* const* d_in, const int* in_sizes, int n_in,
                              void* d_out, int out_size, void* d_ws, size_t ws_size,
                              hipStream_t stream) {
    const float* query     = (const float*)d_in[0];
    const int*   kpm       = (const int*)  d_in[1];
    const float* attn_mask = (const float*)d_in[2];
    const float* attn_bias = (const float*)d_in[3];
    const float* W_in      = (const float*)d_in[4];
    const float* b_in      = (const float*)d_in[5];
    const float* W_out     = (const float*)d_in[6];
    const float* b_out     = (const float*)d_in[7];
    float* out = (float*)d_out;

    // workspace layout (bf16 elements)
    const size_t SEG   = (size_t)64 * TGT * HDIM;       // 4 Mi elems (Q/K/V/attn)
    const size_t N_QRY = (size_t)TGT * BSZ * EMBED;     // 4 Mi
    const size_t N_WIN = (size_t)3 * EMBED * EMBED;     // 3 Mi
    const size_t N_WOUT= (size_t)EMBED * EMBED;         // 1 Mi
    unsigned short* qws    = (unsigned short*)d_ws;
    unsigned short* kws    = qws + SEG;
    unsigned short* vws    = kws + SEG;
    unsigned short* attnws = vws + SEG;
    unsigned short* queryb = attnws + SEG;
    unsigned short* winb   = queryb + N_QRY;
    unsigned short* woutb  = winb + N_WIN;

    // K0: one-shot conversions (8 elems / thread)
    cvt_bf16_kernel<<<(int)(N_QRY  / 8 / 256), 256, 0, stream>>>(query, (unsigned*)queryb, (int)(N_QRY  / 8));
    cvt_bf16_kernel<<<(int)(N_WIN  / 8 / 256), 256, 0, stream>>>(W_in,  (unsigned*)winb,   (int)(N_WIN  / 8));
    cvt_bf16_kernel<<<(int)(N_WOUT / 8 / 256), 256, 0, stream>>>(W_out, (unsigned*)woutb,  (int)(N_WOUT / 8));

    qkv_proj_kernel<<<768, 256, 0, stream>>>(queryb, winb, b_in, qws, kws, vws);
    attn_kernel    <<<512, 256, 0, stream>>>(qws, kws, vws, attn_mask, attn_bias,
                                             kpm, attnws);
    out_proj_kernel<<<256, 256, 0, stream>>>(attnws, woutb, b_out, out);
}
